// MultiHeadAttention_63986422775834
// MI455X (gfx1250) — matrix operand order repaired and checked by reading, not compiled, not run on hardware
//
#include <hip/hip_runtime.h>
#include <hip/hip_bf16.h>

// ---------------------------------------------------------------------------
// Types for CDNA5 WMMA (wave32): v16bf A/B fragments, v8f f32 accumulator.
// ---------------------------------------------------------------------------
typedef __bf16 v16bf __attribute__((ext_vector_type(16)));
typedef float  v8f   __attribute__((ext_vector_type(8)));

union Frag16 {
    v16bf v;
    unsigned short u[16];
    uint4 q[2];
};

// Split fp32 into bf16 hi + bf16 lo correction (hi = RNE(x), lo = RNE(x - hi)).
__device__ __forceinline__ void split_bf16(float x, unsigned short& hi, unsigned short& lo) {
    __bf16 h = (__bf16)x;
    float hf = (float)h;
    __bf16 l = (__bf16)(x - hf);
    hi = __builtin_bit_cast(unsigned short, h);
    lo = __builtin_bit_cast(unsigned short, l);
}

__device__ __forceinline__ unsigned long long pack4(unsigned short a, unsigned short b,
                                                    unsigned short c, unsigned short d) {
    return (unsigned long long)a | ((unsigned long long)b << 16) |
           ((unsigned long long)c << 32) | ((unsigned long long)d << 48);
}

#define BM 128
#define BN 128
#define BK 32
#define LDSS 40 // padded LDS row stride in halves (80B rows, 16B-aligned)

// ---------------------------------------------------------------------------
// Kernel 1: fused QKV projection GEMM, bf16 hi/lo split (3 WMMAs per tile),
// double-buffered LDS with register prefetch of the next K-tile.
//   grid = (Ndim/BN, Mdim/BM, 3), block = 256 (8 waves, 4x2 wave grid).
//   Each wave computes a 32x64 output patch = 2x4 tiles of 16x16.
// ---------------------------------------------------------------------------
__global__ void __launch_bounds__(256)
qkv_gemm_bf16x3(const float* __restrict__ x,
                const float* __restrict__ Wq, const float* __restrict__ Wk,
                const float* __restrict__ Wv,
                const float* __restrict__ bq, const float* __restrict__ bk,
                const float* __restrict__ bv,
                float* __restrict__ qkv)
{
    constexpr int Mdim = 16384, Kdim = 2048, Ndim = 2048;

    __shared__ __align__(16) unsigned short AsH[2][BM * LDSS];
    __shared__ __align__(16) unsigned short AsL[2][BM * LDSS];
    __shared__ __align__(16) unsigned short BsH[2][BN * LDSS];
    __shared__ __align__(16) unsigned short BsL[2][BN * LDSS];

    const int z = blockIdx.z;
    const float* __restrict__ W    = (z == 0) ? Wq : (z == 1) ? Wk : Wv;
    const float* __restrict__ bias = (z == 0) ? bq : (z == 1) ? bk : bv;
    float* __restrict__ out = qkv + (size_t)z * (size_t)Mdim * (size_t)Ndim;

    const int tid   = threadIdx.x;
    const int lane  = tid & 31;
    const int wid   = tid >> 5;
    const int waveM = wid & 3;   // 0..3
    const int waveN = wid >> 2;  // 0..1
    const int bm0 = blockIdx.y * BM;
    const int bn0 = blockIdx.x * BN;

    const int mlane = lane & 15;        // row (A) / col (B) within 16
    const int khalf = (lane >> 4) & 1;  // which K half this lane holds

    // Per-thread global-load ownership (constant across K-steps):
    //  A: float4 slots; slot idx = tid + it*256 -> row idx>>3, cols (idx&7)*4
    //  B: one n column, 4 consecutive k rows; idx = tid + it*256 ->
    //     n = idx&127, k = (idx>>7)*4   (each b32 load is 128B-coalesced/wave)
    float4 aPre[4];
    float  bPre[4][4];

    auto load_globals = [&](int k0) {
#pragma unroll
        for (int it = 0; it < 4; ++it) {
            int idx = tid + it * 256;
            int r   = idx >> 3;
            int c4  = (idx & 7) << 2;
            aPre[it] = *(const float4*)(x + (size_t)(bm0 + r) * Kdim + (k0 + c4));
        }
#pragma unroll
        for (int it = 0; it < 4; ++it) {
            int idx = tid + it * 256;
            int n   = idx & 127;
            int kk  = (idx >> 7) << 2;
#pragma unroll
            for (int i = 0; i < 4; ++i)
                bPre[it][i] = W[(size_t)(k0 + kk + i) * Ndim + (bn0 + n)];
        }
    };

    auto store_lds = [&](int buf) {
#pragma unroll
        for (int it = 0; it < 4; ++it) {
            int idx = tid + it * 256;
            int r   = idx >> 3;
            int c4  = (idx & 7) << 2;
            unsigned short h0, h1, h2, h3, l0, l1, l2, l3;
            split_bf16(aPre[it].x, h0, l0);
            split_bf16(aPre[it].y, h1, l1);
            split_bf16(aPre[it].z, h2, l2);
            split_bf16(aPre[it].w, h3, l3);
            *(unsigned long long*)&AsH[buf][r * LDSS + c4] = pack4(h0, h1, h2, h3);
            *(unsigned long long*)&AsL[buf][r * LDSS + c4] = pack4(l0, l1, l2, l3);
        }
#pragma unroll
        for (int it = 0; it < 4; ++it) {
            int idx = tid + it * 256;
            int n   = idx & 127;
            int kk  = (idx >> 7) << 2;
            unsigned short h0, h1, h2, h3, l0, l1, l2, l3;
            split_bf16(bPre[it][0], h0, l0);
            split_bf16(bPre[it][1], h1, l1);
            split_bf16(bPre[it][2], h2, l2);
            split_bf16(bPre[it][3], h3, l3);
            *(unsigned long long*)&BsH[buf][n * LDSS + kk] = pack4(h0, h1, h2, h3);
            *(unsigned long long*)&BsL[buf][n * LDSS + kk] = pack4(l0, l1, l2, l3);
        }
    };

    v8f acc[2][4];
#pragma unroll
    for (int tm = 0; tm < 2; ++tm)
#pragma unroll
        for (int tn = 0; tn < 4; ++tn)
#pragma unroll
            for (int r = 0; r < 8; ++r)
                acc[tm][tn][r] = 0.0f;

    load_globals(0);

    for (int k0 = 0; k0 < Kdim; k0 += BK) {
        const int buf = (k0 >> 5) & 1;
        store_lds(buf);
        __syncthreads();

        // Prefetch next tile's globals; overlaps the WMMA stream below.
        if (k0 + BK < Kdim) load_globals(k0 + BK);

        // ---- build fragments per ISA 7.12.2 lane maps
        Frag16 aH[2], aL[2], bH[4], bL[4];
#pragma unroll
        for (int tm = 0; tm < 2; ++tm) {
            int m   = waveM * 32 + tm * 16 + mlane;
            int off = m * LDSS + khalf * 8;              // K = khalf*8 + {0..7, 16..23}
            aH[tm].q[0] = *(const uint4*)&AsH[buf][off];
            aH[tm].q[1] = *(const uint4*)&AsH[buf][off + 16];
            aL[tm].q[0] = *(const uint4*)&AsL[buf][off];
            aL[tm].q[1] = *(const uint4*)&AsL[buf][off + 16];
        }
#pragma unroll
        for (int tn = 0; tn < 4; ++tn) {
            int n   = waveN * 64 + tn * 16 + mlane;
            int off = n * LDSS + khalf * 8;              // K = khalf*8 + {0..7, 16..23}
            bH[tn].q[0] = *(const uint4*)&BsH[buf][off];
            bH[tn].q[1] = *(const uint4*)&BsH[buf][off + 16];
            bL[tn].q[0] = *(const uint4*)&BsL[buf][off];
            bL[tn].q[1] = *(const uint4*)&BsL[buf][off + 16];
        }

        // ---- 3-product bf16 emulation of fp32: hi*hi + hi*lo + lo*hi
#pragma unroll
        for (int tm = 0; tm < 2; ++tm)
#pragma unroll
            for (int tn = 0; tn < 4; ++tn) {
                acc[tm][tn] = __builtin_amdgcn_wmma_f32_16x16x32_bf16(
                    false, aH[tm].v, false, bH[tn].v, (short)0, acc[tm][tn], false, false);
                acc[tm][tn] = __builtin_amdgcn_wmma_f32_16x16x32_bf16(
                    false, aH[tm].v, false, bL[tn].v, (short)0, acc[tm][tn], false, false);
                acc[tm][tn] = __builtin_amdgcn_wmma_f32_16x16x32_bf16(
                    false, aL[tm].v, false, bH[tn].v, (short)0, acc[tm][tn], false, false);
            }
        // Single barrier per K-step is sufficient with double buffering:
        // the next store targets buf^1, whose readers were fenced by the
        // barrier in this iteration.
    }

    // ---- epilogue: bias + store (C layout: VGPR r -> row r + khalf*8, col mlane)
#pragma unroll
    for (int tn = 0; tn < 4; ++tn) {
        int n_col = bn0 + waveN * 64 + tn * 16 + mlane;
        float bsc = bias[n_col];
#pragma unroll
        for (int tm = 0; tm < 2; ++tm) {
            int row0 = bm0 + waveM * 32 + tm * 16 + khalf * 8;
#pragma unroll
            for (int r = 0; r < 8; ++r)
                out[(size_t)(row0 + r) * Ndim + n_col] = acc[tm][tn][r] + bsc;
        }
    }
}

// ---------------------------------------------------------------------------
// Kernel 2: row-group attention. One wave32 per group of 8 rows.
//   scores[8][8] = q.k^T / sqrt(D) over full D=2048, softmax, out = w @ v.
//   Pure streaming (memory-bound): all accesses 128B coalesced per wave.
// ---------------------------------------------------------------------------
__global__ void __launch_bounds__(256)
attn_rowgroup(const float* __restrict__ q, const float* __restrict__ k,
              const float* __restrict__ v, float* __restrict__ out)
{
    constexpr int D = 2048;
    const int lane  = threadIdx.x & 31;
    const int wid   = threadIdx.x >> 5;
    const int group = blockIdx.x * 8 + wid;
    const size_t base = (size_t)group * 8 * D;

    float s[8][8];
#pragma unroll
    for (int h = 0; h < 8; ++h)
#pragma unroll
        for (int g = 0; g < 8; ++g) s[h][g] = 0.0f;

    for (int j = 0; j < D / 32; ++j) {
        const int d = lane + (j << 5);
        float qv[8], kv[8];
#pragma unroll
        for (int h = 0; h < 8; ++h) {
            qv[h] = q[base + (size_t)h * D + d];
            kv[h] = k[base + (size_t)h * D + d];
        }
#pragma unroll
        for (int h = 0; h < 8; ++h)
#pragma unroll
            for (int g = 0; g < 8; ++g)
                s[h][g] = fmaf(qv[h], kv[g], s[h][g]);
    }

    // butterfly reduce across the 32 lanes; every lane ends with full sums
#pragma unroll
    for (int h = 0; h < 8; ++h)
#pragma unroll
        for (int g = 0; g < 8; ++g) {
            float t = s[h][g];
#pragma unroll
            for (int off = 16; off >= 1; off >>= 1)
                t += __shfl_xor(t, off, 32);
            s[h][g] = t;
        }

    const float scale = 0.022097086912079608f; // 1/sqrt(2048)
#pragma unroll
    for (int h = 0; h < 8; ++h) {
        float m = -3.4e38f;
#pragma unroll
        for (int g = 0; g < 8; ++g) { s[h][g] *= scale; m = fmaxf(m, s[h][g]); }
        float sum = 0.0f;
#pragma unroll
        for (int g = 0; g < 8; ++g) { s[h][g] = __expf(s[h][g] - m); sum += s[h][g]; }
        float inv = 1.0f / sum;
#pragma unroll
        for (int g = 0; g < 8; ++g) s[h][g] *= inv;
    }

    for (int j = 0; j < D / 32; ++j) {
        const int d = lane + (j << 5);
        float vv[8];
#pragma unroll
        for (int g = 0; g < 8; ++g) vv[g] = v[base + (size_t)g * D + d];
#pragma unroll
        for (int h = 0; h < 8; ++h) {
            float o = 0.0f;
#pragma unroll
            for (int g = 0; g < 8; ++g) o = fmaf(s[h][g], vv[g], o);
            out[base + (size_t)h * D + d] = o;
        }
    }
}

// ---------------------------------------------------------------------------
// Host launch. Workspace holds Q,K,V planes: 3 * 16384 * 2048 * 4B = 403 MB.
// ---------------------------------------------------------------------------
extern "C" void kernel_launch(void* const* d_in, const int* in_sizes, int n_in,
                              void* d_out, int out_size, void* d_ws, size_t ws_size,
                              hipStream_t stream) {
    const float* x  = (const float*)d_in[0];
    const float* Wq = (const float*)d_in[1];
    const float* bq = (const float*)d_in[2];
    const float* Wk = (const float*)d_in[3];
    const float* bk = (const float*)d_in[4];
    const float* Wv = (const float*)d_in[5];
    const float* bv = (const float*)d_in[6];
    float* out = (float*)d_out;

    constexpr int M = 16384, D = 2048;
    float* q = (float*)d_ws;
    float* k = q + (size_t)M * D;
    float* v = k + (size_t)M * D;

    dim3 grid(D / BN, M / BM, 3);
    hipLaunchKernelGGL(qkv_gemm_bf16x3, grid, dim3(256), 0, stream,
                       x, Wq, Wk, Wv, bq, bk, bv, (float*)d_ws);

    // 2048 groups of 8 rows; 8 waves per block -> 256 blocks
    hipLaunchKernelGGL(attn_rowgroup, dim3(M / 8 / 8), dim3(256), 0, stream,
                       q, k, v, out);

    (void)in_sizes; (void)n_in; (void)out_size; (void)ws_size;
}
